// BertBiAttention_7172595384430
// MI455X (gfx1250) — matrix operand order repaired and checked by reading, not compiled, not run on hardware
//
#include <hip/hip_runtime.h>
#include <hip/hip_bf16.h>

// ---------------------------------------------------------------------------
// BERT bi-attention on gfx1250:
//   - all GEMMs (6 projections, QK^T, P*V) on v_wmma_f32_16x16x32_bf16
//   - attention B-operands staged block-wide via global_load_async_to_lds_b128
//     (ASYNCcnt + s_wait_asynccnt), fragments read back as ds_load_b128
// ---------------------------------------------------------------------------

typedef __attribute__((ext_vector_type(16))) __bf16 v16bf;
typedef __attribute__((ext_vector_type(8)))  float  v8f;

#define Bz   16
#define S1z  256
#define S2z  512
#define VHID 1024
#define THID 768
#define BIH  1024
#define NH   8
#define DH   128

__device__ __forceinline__ unsigned short f2bf(float f) {
    union { float f; unsigned u; } v; v.f = f;
    unsigned u = v.u;
    unsigned r = (u + 0x7FFFu + ((u >> 16) & 1u)) >> 16;   // RNE
    return (unsigned short)r;
}

union FragU { v16bf v; uint4 q[2]; };

// A fragment 16x32 bf16: lane l -> row m=l&15, K = (l>>4)*8 .. +7 and +16..+23
__device__ __forceinline__ v16bf load_a_frag(const unsigned short* base, int stride, int kpos) {
    int lane = threadIdx.x & 31;
    int m    = lane & 15;
    int koff = (lane >> 4) * 8;
    const unsigned short* p = base + (size_t)m * stride + kpos + koff;
    FragU u;
    u.q[0] = *reinterpret_cast<const uint4*>(p);
    u.q[1] = *reinterpret_cast<const uint4*>(p + 16);
    return u.v;
}

// B fragment 32x16 bf16 from an [N,K] layout: lane l -> col n=l&15,
// K = (l>>4)*8 .. +7 and +16..+23
__device__ __forceinline__ v16bf load_b_frag(const unsigned short* base, int stride, int kpos) {
    int lane = threadIdx.x & 31;
    int n    = lane & 15;
    int koff = (lane >> 4) * 8;
    const unsigned short* p = base + (size_t)n * stride + kpos + koff;
    FragU u;
    u.q[0] = *reinterpret_cast<const uint4*>(p);
    u.q[1] = *reinterpret_cast<const uint4*>(p + 16);
    return u.v;
}

// Async copy of one 16-byte chunk global -> LDS (per-lane addresses).
__device__ __forceinline__ void async_g2l_b128(const unsigned short* g, unsigned short* l) {
    unsigned           lds  = (unsigned)(unsigned long long)l;           // LDS byte address
    unsigned long long gadr = (unsigned long long)g;
    asm volatile("global_load_async_to_lds_b128 %0, %1, off"
                 :: "v"(lds), "v"(gadr) : "memory");
}

__device__ __forceinline__ void wait_async_and_barrier() {
    asm volatile("s_wait_asynccnt 0x0" ::: "memory");
    __syncthreads();
}

// ---------------------------------------------------------------------------
// Conversion kernels
// ---------------------------------------------------------------------------
__global__ void cvt_weight_t_kernel(const float* __restrict__ W,
                                    unsigned short* __restrict__ Wt, int K, int N) {
    int idx = blockIdx.x * 256 + threadIdx.x;
    if (idx >= K * N) return;
    int k = idx / N, n = idx % N;
    Wt[(size_t)n * K + k] = f2bf(W[idx]);
}

__global__ void cvt_act_kernel(const float* __restrict__ X,
                               unsigned short* __restrict__ Y, int n) {
    int idx = blockIdx.x * 256 + threadIdx.x;
    if (idx < n) Y[idx] = f2bf(X[idx]);
}

// ---------------------------------------------------------------------------
// Projection GEMM: C[m,n] = sum_k X[m,k]*Wt[n,k] + bias[n]
//   Each wave: 32(M) x 64(N) -> 2 A-tiles share each B fragment (8 WMMA/k-step).
//   Block = 8 waves -> 256(M) x 64(N).
//   MODE 0: store bf16 row-major [M,N]  (== [b,s,h,d] since N = H*D)
//   MODE 1: store bf16 transposed-V layout [(b*H+h)*D + d][S]
// ---------------------------------------------------------------------------
template<int MODE>
__global__ void proj_gemm_kernel(const unsigned short* __restrict__ X,
                                 const unsigned short* __restrict__ Wt,
                                 const float* __restrict__ bias,
                                 unsigned short* __restrict__ out,
                                 int M, int N, int K, int S) {
    int wave = threadIdx.x >> 5;
    int lane = threadIdx.x & 31;
    int m0 = (blockIdx.y * 8 + wave) * 32;
    int n0 = blockIdx.x * 64;

    v8f acc0[4] = {};
    v8f acc1[4] = {};
    const unsigned short* Abase0 = X + (size_t)m0 * K;
    const unsigned short* Abase1 = X + (size_t)(m0 + 16) * K;
    for (int kp = 0; kp < K; kp += 32) {
        v16bf a0 = load_a_frag(Abase0, K, kp);
        v16bf a1 = load_a_frag(Abase1, K, kp);
#pragma unroll
        for (int t = 0; t < 4; ++t) {
            v16bf b = load_b_frag(Wt + (size_t)(n0 + t * 16) * K, K, kp);
            acc0[t] = __builtin_amdgcn_wmma_f32_16x16x32_bf16(
                false, a0, false, b, (short)0, acc0[t], false, false);
            acc1[t] = __builtin_amdgcn_wmma_f32_16x16x32_bf16(
                false, a1, false, b, (short)0, acc1[t], false, false);
        }
    }

    int n_lane = lane & 15;
    int r_base = (lane >> 4) * 8;
#pragma unroll
    for (int half = 0; half < 2; ++half) {
        v8f* acc = half ? acc1 : acc0;
        int mh = m0 + half * 16;
#pragma unroll
        for (int t = 0; t < 4; ++t) {
            int n = n0 + t * 16 + n_lane;
            float bv = bias[n];
#pragma unroll
            for (int r = 0; r < 8; ++r) {
                int m = mh + r_base + r;
                float val = acc[t][r] + bv;
                if (MODE == 0) {
                    out[(size_t)m * N + n] = f2bf(val);
                } else {
                    int b = m / S, s = m % S;
                    int h = n >> 7, d = n & 127;
                    out[(((size_t)b * NH + h) * DH + d) * S + s] = f2bf(val);
                }
            }
        }
    }
}

// ---------------------------------------------------------------------------
// Scores: scores[b,h,q,k] = scale * <Q[b,q,h,:], K[b,k,h,:]> + mask[b,k]
// The 64-key x 128-d K-strip is shared by all 8 waves -> async-stage it in LDS.
// ---------------------------------------------------------------------------
__global__ void scores_kernel(const unsigned short* __restrict__ Q,
                              const unsigned short* __restrict__ Kmat,
                              const float* __restrict__ mask,      // [B, Sk]
                              float* __restrict__ scores,
                              int Sq, int Sk, float scale) {
    __shared__ unsigned short skl[64 * DH];   // [key][d], 16 KB

    int bh = blockIdx.z;
    int b = bh >> 3, h = bh & 7;
    int wave = threadIdx.x >> 5;
    int lane = threadIdx.x & 31;
    int q0 = (blockIdx.y * 8 + wave) * 16;
    int k0 = blockIdx.x * 64;

    // Cooperative async stage: 64 keys x 128 d x 2B = 1024 chunks of 16B.
#pragma unroll
    for (int j = 0; j < 4; ++j) {
        int c = threadIdx.x + 256 * j;          // 0..1023
        int key = c >> 4, part = c & 15;        // 16 chunks per key row
        const unsigned short* g =
            Kmat + (size_t)b * Sk * BIH + (size_t)(k0 + key) * BIH + h * DH + part * 8;
        async_g2l_b128(g, skl + c * 8);
    }
    wait_async_and_barrier();

    const unsigned short* Qbase = Q + (size_t)b * Sq * BIH + h * DH + (size_t)q0 * BIH;
    v8f acc[4] = {};
    for (int kp = 0; kp < DH; kp += 32) {
        v16bf a = load_a_frag(Qbase, BIH, kp);
#pragma unroll
        for (int t = 0; t < 4; ++t) {
            v16bf bb = load_b_frag(skl + (size_t)(t * 16) * DH, DH, kp);
            acc[t] = __builtin_amdgcn_wmma_f32_16x16x32_bf16(
                false, a, false, bb, (short)0, acc[t], false, false);
        }
    }

    int n_lane = lane & 15;
    int r_base = (lane >> 4) * 8;
    float* srow = scores + ((size_t)bh * Sq + q0) * Sk + k0;
#pragma unroll
    for (int t = 0; t < 4; ++t) {
        int n = t * 16 + n_lane;
        float mv = mask[(size_t)b * Sk + k0 + n];
#pragma unroll
        for (int r = 0; r < 8; ++r)
            srow[(size_t)(r_base + r) * Sk + n] = acc[t][r] * scale + mv;
    }
}

// ---------------------------------------------------------------------------
// Row softmax: f32 scores -> bf16 probs. One wave32 per row.
// ---------------------------------------------------------------------------
__global__ void softmax_kernel(const float* __restrict__ scores,
                               unsigned short* __restrict__ probs, int cols) {
    int row  = blockIdx.x * 8 + (threadIdx.x >> 5);
    int lane = threadIdx.x & 31;
    const float* src = scores + (size_t)row * cols;
    unsigned short* dst = probs + (size_t)row * cols;

    float m = -1e30f;
    for (int c = lane; c < cols; c += 32) m = fmaxf(m, src[c]);
#pragma unroll
    for (int o = 16; o > 0; o >>= 1) m = fmaxf(m, __shfl_xor(m, o, 32));

    float s = 0.f;
    for (int c = lane; c < cols; c += 32) s += __expf(src[c] - m);
#pragma unroll
    for (int o = 16; o > 0; o >>= 1) s += __shfl_xor(s, o, 32);

    float inv = 1.0f / s;
    for (int c = lane; c < cols; c += 32)
        dst[c] = f2bf(__expf(src[c] - m) * inv);
}

// ---------------------------------------------------------------------------
// Context: out[b,q,h*D+d] = sum_key P[bh,q,key] * Vt[bh,d,key]
// The 64-d x Sk Vt-strip is shared by all 8 waves -> async-stage it in LDS
// (dynamic LDS: 64*Sk*2 bytes = 32..64 KB).
// ---------------------------------------------------------------------------
__global__ void ctx_kernel(const unsigned short* __restrict__ P,
                           const unsigned short* __restrict__ Vt,
                           float* __restrict__ out, int Sq, int Sk) {
    extern __shared__ unsigned short svl[];   // [64 d rows][Sk keys]

    int bh = blockIdx.z;
    int b = bh >> 3, h = bh & 7;
    int wave = threadIdx.x >> 5;
    int lane = threadIdx.x & 31;
    int q0 = (blockIdx.y * 8 + wave) * 16;
    int d0 = blockIdx.x * 64;

    // Cooperative async stage: 64 * Sk * 2 bytes -> 8*Sk chunks of 16B.
    int chunks_per_row = Sk >> 3;             // 16B chunks per d-row
    int per_thread     = Sk >> 5;             // (64*Sk/8)/256
    for (int j = 0; j < per_thread; ++j) {
        int c = threadIdx.x + 256 * j;
        int drow = c / chunks_per_row;
        int part = c % chunks_per_row;
        const unsigned short* g =
            Vt + ((size_t)bh * DH + d0 + drow) * Sk + part * 8;
        async_g2l_b128(g, svl + c * 8);
    }
    wait_async_and_barrier();

    const unsigned short* Pbase = P + ((size_t)bh * Sq + q0) * Sk;
    v8f acc[4] = {};
    for (int kp = 0; kp < Sk; kp += 32) {
        v16bf a = load_a_frag(Pbase, Sk, kp);
#pragma unroll
        for (int t = 0; t < 4; ++t) {
            v16bf bb = load_b_frag(svl + (size_t)(t * 16) * Sk, Sk, kp);
            acc[t] = __builtin_amdgcn_wmma_f32_16x16x32_bf16(
                false, a, false, bb, (short)0, acc[t], false, false);
        }
    }

    int n_lane = lane & 15;
    int r_base = (lane >> 4) * 8;
#pragma unroll
    for (int t = 0; t < 4; ++t) {
        int d = d0 + t * 16 + n_lane;
#pragma unroll
        for (int r = 0; r < 8; ++r) {
            int q = q0 + r_base + r;
            out[((size_t)b * Sq + q) * BIH + h * DH + d] = acc[t][r];
        }
    }
}

// ---------------------------------------------------------------------------
// Launch
// ---------------------------------------------------------------------------
extern "C" void kernel_launch(void* const* d_in, const int* in_sizes, int n_in,
                              void* d_out, int out_size, void* d_ws, size_t ws_size,
                              hipStream_t stream) {
    const float* x1    = (const float*)d_in[0];
    const float* mask1 = (const float*)d_in[1];
    const float* x2    = (const float*)d_in[2];
    const float* mask2 = (const float*)d_in[3];
    const float* W[6]  = { (const float*)d_in[4],  (const float*)d_in[6],
                           (const float*)d_in[8],  (const float*)d_in[10],
                           (const float*)d_in[12], (const float*)d_in[14] };
    const float* bias[6] = { (const float*)d_in[5],  (const float*)d_in[7],
                             (const float*)d_in[9],  (const float*)d_in[11],
                             (const float*)d_in[13], (const float*)d_in[15] };
    float* out = (float*)d_out;

    const int M1 = Bz * S1z;            // 4096
    const int M2 = Bz * S2z;            // 8192

    // Scratch layout (bf16 stored as unsigned short; scores f32)
    char* ws = (char*)d_ws;
    size_t off = 0;
    auto alloc = [&](size_t bytes) { size_t o = off; off = (off + bytes + 255) & ~(size_t)255; return o; };

    size_t oWt[6];
    oWt[0] = alloc((size_t)VHID * BIH * 2);   // Wq1t
    oWt[1] = alloc((size_t)VHID * BIH * 2);   // Wk1t
    oWt[2] = alloc((size_t)VHID * BIH * 2);   // Wv1t
    oWt[3] = alloc((size_t)THID * BIH * 2);   // Wq2t
    oWt[4] = alloc((size_t)THID * BIH * 2);   // Wk2t
    oWt[5] = alloc((size_t)THID * BIH * 2);   // Wv2t
    size_t oX1  = alloc((size_t)M1 * VHID * 2);
    size_t oX2  = alloc((size_t)M2 * THID * 2);
    size_t oQ1  = alloc((size_t)M1 * BIH * 2);
    size_t oK1  = alloc((size_t)M1 * BIH * 2);
    size_t oV1t = alloc((size_t)M1 * BIH * 2);
    size_t oQ2  = alloc((size_t)M2 * BIH * 2);
    size_t oK2  = alloc((size_t)M2 * BIH * 2);
    size_t oV2t = alloc((size_t)M2 * BIH * 2);
    size_t oSc  = alloc((size_t)Bz * NH * S2z * S1z * 4);   // reused by both attns
    size_t oPr  = alloc((size_t)Bz * NH * S2z * S1z * 2);
    (void)ws_size;

    unsigned short* wt[6]; for (int i = 0; i < 6; ++i) wt[i] = (unsigned short*)(ws + oWt[i]);
    unsigned short* x1b = (unsigned short*)(ws + oX1);
    unsigned short* x2b = (unsigned short*)(ws + oX2);
    unsigned short* q1  = (unsigned short*)(ws + oQ1);
    unsigned short* k1  = (unsigned short*)(ws + oK1);
    unsigned short* v1t = (unsigned short*)(ws + oV1t);
    unsigned short* q2  = (unsigned short*)(ws + oQ2);
    unsigned short* k2  = (unsigned short*)(ws + oK2);
    unsigned short* v2t = (unsigned short*)(ws + oV2t);
    float*          sc  = (float*)(ws + oSc);
    unsigned short* pr  = (unsigned short*)(ws + oPr);

    const float scale = 0.088388347648318447f;   // 1/sqrt(128)

    // --- conversions ---
    cvt_act_kernel<<<(M1 * VHID + 255) / 256, 256, 0, stream>>>(x1, x1b, M1 * VHID);
    cvt_act_kernel<<<(M2 * THID + 255) / 256, 256, 0, stream>>>(x2, x2b, M2 * THID);
    for (int i = 0; i < 3; ++i)
        cvt_weight_t_kernel<<<(VHID * BIH + 255) / 256, 256, 0, stream>>>(W[i], wt[i], VHID, BIH);
    for (int i = 3; i < 6; ++i)
        cvt_weight_t_kernel<<<(THID * BIH + 255) / 256, 256, 0, stream>>>(W[i], wt[i], THID, BIH);

    // --- projections (WMMA, 32x64 per wave) ---
    dim3 blk(256);
    dim3 g1(BIH / 64, M1 / 256);
    dim3 g2(BIH / 64, M2 / 256);
    proj_gemm_kernel<0><<<g1, blk, 0, stream>>>(x1b, wt[0], bias[0], q1,  M1, BIH, VHID, S1z);
    proj_gemm_kernel<0><<<g1, blk, 0, stream>>>(x1b, wt[1], bias[1], k1,  M1, BIH, VHID, S1z);
    proj_gemm_kernel<1><<<g1, blk, 0, stream>>>(x1b, wt[2], bias[2], v1t, M1, BIH, VHID, S1z);
    proj_gemm_kernel<0><<<g2, blk, 0, stream>>>(x2b, wt[3], bias[3], q2,  M2, BIH, THID, S2z);
    proj_gemm_kernel<0><<<g2, blk, 0, stream>>>(x2b, wt[4], bias[4], k2,  M2, BIH, THID, S2z);
    proj_gemm_kernel<1><<<g2, blk, 0, stream>>>(x2b, wt[5], bias[5], v2t, M2, BIH, THID, S2z);

    // --- attention 1: text queries (S2) over vision keys (S1) -> context1 ---
    {
        dim3 gs(S1z / 64, S2z / 128, Bz * NH);
        scores_kernel<<<gs, blk, 0, stream>>>(q2, k1, mask1, sc, S2z, S1z, scale);
        int rows = Bz * NH * S2z;
        softmax_kernel<<<rows / 8, blk, 0, stream>>>(sc, pr, S1z);
        dim3 gc(DH / 64, S2z / 128, Bz * NH);
        ctx_kernel<<<gc, blk, (size_t)64 * S1z * 2, stream>>>(pr, v1t, out, S2z, S1z);
    }

    // --- attention 2: vision queries (S1) over text keys (S2) -> context2 ---
    {
        float* out2 = out + (size_t)Bz * S2z * BIH;
        dim3 gs(S2z / 64, S1z / 128, Bz * NH);
        scores_kernel<<<gs, blk, 0, stream>>>(q1, k2, mask2, sc, S1z, S2z, scale);
        int rows = Bz * NH * S1z;
        softmax_kernel<<<rows / 8, blk, 0, stream>>>(sc, pr, S2z);
        dim3 gc(DH / 64, S1z / 128, Bz * NH);
        ctx_kernel<<<gc, blk, (size_t)64 * S2z * 2, stream>>>(pr, v2t, out2, S1z, S2z);
    }
}
